// BitLinear_87952340287684
// MI455X (gfx1250) — matrix operand order repaired and checked by reading, not compiled, not run on hardware
//
#include <hip/hip_runtime.h>
#include <hip/hip_bf16.h>

// ---------------------------------------------------------------------------
// BitLinear on MI455X (gfx1250): LN+quant -> f16, w -> f16 transposed, then
// WMMA F16->F32 GEMM with 128x128x64 block tiles, double-buffered LDS filled
// by CDNA5 async global->LDS DMA (global_load_async_to_lds_b128 / ASYNCcnt).
// M=8192, K=4096, N=4096 -> ~274 GFLOP, compute-bound on the matrix pipe.
// ---------------------------------------------------------------------------

typedef _Float16 v16h __attribute__((ext_vector_type(16)));
typedef _Float16 v8h  __attribute__((ext_vector_type(8)));
typedef _Float16 v4h  __attribute__((ext_vector_type(4)));
typedef float    v8f  __attribute__((ext_vector_type(8)));

#define N_IN  4096
#define N_OUT 4096
#define M_TOT 8192
#define QB    128.0f
#define EP    0.01f
#define LN_EPS 1e-5f

// ---------------------------------------------------------------------------
// Kernel 1: fused LayerNorm + absmax quantize + cast to f16.
// One block (256 thr = 8 waves) per row of 4096.
// ---------------------------------------------------------------------------
__global__ __launch_bounds__(256) void ln_quant_kernel(
    const float* __restrict__ x, const float* __restrict__ lng,
    const float* __restrict__ lnb, const float* __restrict__ gamma_p,
    _Float16* __restrict__ xq)
{
    const int row = blockIdx.x;
    const int t   = threadIdx.x;
    const float* xr = x + (size_t)row * N_IN;

    float4 v[4];
    float s = 0.f, s2 = 0.f;
#pragma unroll
    for (int i = 0; i < 4; ++i) {
        v[i] = ((const float4*)xr)[t * 4 + i];
        s  += v[i].x + v[i].y + v[i].z + v[i].w;
        s2 += v[i].x * v[i].x + v[i].y * v[i].y + v[i].z * v[i].z + v[i].w * v[i].w;
    }
#pragma unroll
    for (int off = 16; off > 0; off >>= 1) {
        s  += __shfl_xor(s,  off, 32);
        s2 += __shfl_xor(s2, off, 32);
    }
    __shared__ float red1[8], red2[8];
    __shared__ float mu_s, rstd_s;
    const int wave = t >> 5, lane = t & 31;
    if (lane == 0) { red1[wave] = s; red2[wave] = s2; }
    __syncthreads();
    if (t == 0) {
        float S = 0.f, S2 = 0.f;
#pragma unroll
        for (int i = 0; i < 8; ++i) { S += red1[i]; S2 += red2[i]; }
        const float mu  = S * (1.0f / N_IN);
        const float var = S2 * (1.0f / N_IN) - mu * mu;
        mu_s   = mu;
        rstd_s = rsqrtf(var + LN_EPS);
    }
    __syncthreads();
    const float mu = mu_s, rstd = rstd_s;
    const float qs = QB / gamma_p[0];
    const float lo = -QB + EP, hi = QB - EP;

    _Float16* outr = xq + (size_t)row * N_IN;
#pragma unroll
    for (int i = 0; i < 4; ++i) {
        const int idx = t * 16 + i * 4;
        const float4 g4 = ((const float4*)lng)[idx >> 2];
        const float4 b4 = ((const float4*)lnb)[idx >> 2];
        float e0 = ((v[i].x - mu) * rstd * g4.x + b4.x) * qs;
        float e1 = ((v[i].y - mu) * rstd * g4.y + b4.y) * qs;
        float e2 = ((v[i].z - mu) * rstd * g4.z + b4.z) * qs;
        float e3 = ((v[i].w - mu) * rstd * g4.w + b4.w) * qs;
        e0 = fminf(fmaxf(e0, lo), hi);
        e1 = fminf(fmaxf(e1, lo), hi);
        e2 = fminf(fmaxf(e2, lo), hi);
        e3 = fminf(fmaxf(e3, lo), hi);
        v4h h; h[0] = (_Float16)e0; h[1] = (_Float16)e1; h[2] = (_Float16)e2; h[3] = (_Float16)e3;
        *(v4h*)(outr + idx) = h;
    }
}

// ---------------------------------------------------------------------------
// Kernel 2: one-time weight cast + transpose: wT[n][k] = (f16) w[k][n]
// ---------------------------------------------------------------------------
__global__ __launch_bounds__(256) void w_transpose_kernel(
    const float* __restrict__ w, _Float16* __restrict__ wt)
{
    __shared__ _Float16 tile[32][33];
    const int n0 = blockIdx.x * 32;
    const int k0 = blockIdx.y * 32;
    const int tx = threadIdx.x;   // 0..31
    const int ty = threadIdx.y;   // 0..7
#pragma unroll
    for (int j = 0; j < 32; j += 8)
        tile[ty + j][tx] = (_Float16)w[(size_t)(k0 + ty + j) * N_OUT + n0 + tx];
    __syncthreads();
#pragma unroll
    for (int j = 0; j < 32; j += 8)
        wt[(size_t)(n0 + ty + j) * N_IN + k0 + tx] = tile[tx][ty + j];
}

// ---------------------------------------------------------------------------
// Kernel 3: WMMA GEMM with async-LDS double buffering.
// Block tile 128(M) x 128(N), K-tile 64 (2 wmma K-steps of 32).
// 8 waves: 4 along M x 2 along N; wave tile 32x64 = 2x4 accumulators.
// ---------------------------------------------------------------------------
#define TK  64
#define STR 72   // halves per LDS row: 64 + 8 pad (144B, 16B-aligned)

static __device__ __forceinline__ v16h cat8(v8h lo, v8h hi) {
    return __builtin_shufflevector(lo, hi, 0,1,2,3,4,5,6,7,8,9,10,11,12,13,14,15);
}

// CDNA5 async DMA: global -> LDS, 16B per lane, tracked by ASYNCcnt.
static __device__ __forceinline__ void async_ld16(unsigned lds_addr, const _Float16* g) {
    asm volatile("global_load_async_to_lds_b128 %0, %1, off"
                 :: "v"(lds_addr), "v"(g) : "memory");
}
static __device__ __forceinline__ void wait_async0(void) {
    asm volatile("s_wait_asynccnt 0x0" ::: "memory");
}

__global__ __launch_bounds__(256, 2) void gemm_wmma_kernel(
    const _Float16* __restrict__ A, const _Float16* __restrict__ Bt,
    float* __restrict__ C,
    const float* __restrict__ beta_p, const float* __restrict__ gamma_p)
{
    __shared__ __align__(16) _Float16 sA[2][128 * STR];
    __shared__ __align__(16) _Float16 sB[2][128 * STR];

    const int t    = threadIdx.x;
    const int wave = t >> 5;
    const int lane = t & 31;
    const int lrow = lane & 15;
    const int hiH  = lane >> 4;          // 0: lanes 0-15, 1: lanes 16-31
    const int wm   = wave & 3;           // M offset wm*32
    const int wn   = wave >> 2;          // N offset wn*64
    const int m0   = blockIdx.y * 128;
    const int n0   = blockIdx.x * 128;

    // LDS byte addresses for the async DMA destinations (low 32 bits of the
    // flat pointer hold the LDS offset per the aperture mapping).
    const unsigned ldsA0 = (unsigned)(size_t)&sA[0][0];
    const unsigned ldsB0 = (unsigned)(size_t)&sB[0][0];
    const unsigned bufBytes = 128 * STR * 2;

    v8f acc[2][4] = {};

    // Tile fill: 128 rows x 64 halves = 1024 x 16B chunks; 4 per thread per mat.
    auto issue_tile = [&](int kt, int b) {
        const unsigned aB = ldsA0 + (unsigned)b * bufBytes;
        const unsigned bB = ldsB0 + (unsigned)b * bufBytes;
#pragma unroll
        for (int q = 0; q < 4; ++q) {
            const int chunk = t + q * 256;
            const int row = chunk >> 3, cc = chunk & 7;
            const unsigned loff = (unsigned)(row * STR + cc * 8) * 2u;
            async_ld16(aB + loff, A  + (size_t)(m0 + row) * N_IN + kt + cc * 8);
            async_ld16(bB + loff, Bt + (size_t)(n0 + row) * N_IN + kt + cc * 8);
        }
    };

    issue_tile(0, 0);          // prologue fill of buffer 0
    int buf = 0;

    for (int kt = 0; kt < N_IN; kt += TK) {
        wait_async0();          // our fills for `buf` have landed in LDS
        __syncthreads();        // everyone's fills landed; everyone done reading buf^1
        if (kt + TK < N_IN)
            issue_tile(kt + TK, buf ^ 1);   // overlap next fill with WMMA work

        const _Float16* pA = sA[buf];
        const _Float16* pB = sB[buf];
#pragma unroll
        for (int s = 0; s < 2; ++s) {       // two 16x16x32 K-steps per tile
            const int ks = s * 32;
            // A frag (16x32): lanes 0-15 K=+0-7,+16-23 ; lanes 16-31 K=+8-15,+24-31
            v16h af[2];
#pragma unroll
            for (int i = 0; i < 2; ++i) {
                const int r = wm * 32 + i * 16 + lrow;
                const v8h lo = *(const v8h*)(pA + r * STR + ks + hiH * 8);
                const v8h hi = *(const v8h*)(pA + r * STR + ks + hiH * 8 + 16);
                af[i] = cat8(lo, hi);
            }
            // B frag (32x16, col per lane): lanes 0-15 K=+0-7,+16-23 ; 16-31 K=+8-15,+24-31
            v16h bf[4];
#pragma unroll
            for (int j = 0; j < 4; ++j) {
                const int n = wn * 64 + j * 16 + lrow;
                const v8h lo = *(const v8h*)(pB + n * STR + ks + hiH * 8);
                const v8h hi = *(const v8h*)(pB + n * STR + ks + hiH * 8 + 16);
                bf[j] = cat8(lo, hi);
            }
#pragma unroll
            for (int i = 0; i < 2; ++i)
#pragma unroll
                for (int j = 0; j < 4; ++j)
                    acc[i][j] = __builtin_amdgcn_wmma_f32_16x16x32_f16(
                        false, af[i], false, bf[j], (short)0, acc[i][j], false, false);
        }
        buf ^= 1;
    }

    const float scale = beta_p[0] * gamma_p[0] * (1.0f / QB);
#pragma unroll
    for (int i = 0; i < 2; ++i) {
#pragma unroll
        for (int j = 0; j < 4; ++j) {
            const int col = n0 + wn * 64 + j * 16 + lrow;
#pragma unroll
            for (int r = 0; r < 8; ++r) {
                const int row = m0 + wm * 32 + i * 16 + hiH * 8 + r;
                C[(size_t)row * N_OUT + col] = acc[i][j][r] * scale;
            }
        }
    }
}

// ---------------------------------------------------------------------------
// Launch: inputs = { x, w, ln_gamma, ln_beta, beta, gamma }
// ws layout: [ xq f16 8192*4096 | wT f16 4096*4096 ]  (96 MB)
// ---------------------------------------------------------------------------
extern "C" void kernel_launch(void* const* d_in, const int* in_sizes, int n_in,
                              void* d_out, int out_size, void* d_ws, size_t ws_size,
                              hipStream_t stream) {
    const float* x     = (const float*)d_in[0];
    const float* w     = (const float*)d_in[1];
    const float* lng   = (const float*)d_in[2];
    const float* lnb   = (const float*)d_in[3];
    const float* betap = (const float*)d_in[4];
    const float* gammp = (const float*)d_in[5];
    float* out = (float*)d_out;

    _Float16* xq = (_Float16*)d_ws;
    _Float16* wt = xq + (size_t)M_TOT * N_IN;

    ln_quant_kernel<<<M_TOT, 256, 0, stream>>>(x, lng, lnb, gammp, xq);
    w_transpose_kernel<<<dim3(N_OUT / 32, N_IN / 32), dim3(32, 8), 0, stream>>>(w, wt);
    gemm_wmma_kernel<<<dim3(N_OUT / 128, M_TOT / 128), 256, 0, stream>>>(xq, wt, out, betap, gammp);
}
